// SlotSequenceGenerator_47364899340350
// MI455X (gfx1250) — matrix operand order repaired and checked by reading, not compiled, not run on hardware
//
#include <hip/hip_runtime.h>

typedef __bf16 bf16;
typedef __attribute__((ext_vector_type(8)))  __bf16 bf16x8;
typedef __attribute__((ext_vector_type(16))) __bf16 bf16x16;
typedef __attribute__((ext_vector_type(8)))  float  f32x8;

#define NSLOT 7
#define BSZ   8192
#define DIN   1024
#define EEMB  256
#define HDIM  512
#define SHDIM 256
#define VDIM  100
#define D0    (DIN + EEMB)    /* 1280 */
#define XHW   (D0 + HDIM)     /* 1792 */
#define CHW   (2 * HDIM)      /* 1024 : [cur | h1] */
#define G4H   (4 * HDIM)      /* 2048 */

/* ---------------- GEMM: C[M,N] = A[M,K] * W[N,K]^T (+bias) ---------------- */
#define BM 128
#define BN 128
#define BK 32
#define STRD 40   /* padded LDS row stride in bf16 elements (80B) */
#define LTILE (BM * STRD)
#define NSTAGE 3  /* async pipeline depth: 2 tiles in flight */

#define FLAG_RELU 2

__device__ __forceinline__ float sigmoidf_(float x) { return 1.0f / (1.0f + __expf(-x)); }

/* CDNA5 async DMA: global -> LDS, 16B per lane, tracked by ASYNCcnt */
__device__ __forceinline__ void async_b128(unsigned lds_off, const bf16* g) {
  asm volatile("global_load_async_to_lds_b128 %0, %1, off"
               :: "v"(lds_off), "v"(g) : "memory");
}
/* async loads complete in order: <=4 outstanding proves the older tile done */
__device__ __forceinline__ void wait_async4() {
  asm volatile("s_wait_asynccnt 0x4" ::: "memory");
}
__device__ __forceinline__ void wait_async0() {
  asm volatile("s_wait_asynccnt 0x0" ::: "memory");
}

__global__ __launch_bounds__(256) void gemm_bf16_nt(
    const bf16* __restrict__ A, int lda,
    const bf16* __restrict__ W, int ldw,
    float* __restrict__ C, int ldc,
    int N, int K,
    const float* __restrict__ bias0,
    const float* __restrict__ bias1,
    int flags)
{
  __shared__ bf16 As[NSTAGE][LTILE];
  __shared__ bf16 Ws[NSTAGE][LTILE];

  const int tid   = threadIdx.x;
  const int lane  = tid & 31;
  const int wave  = tid >> 5;
  const int wm    = wave >> 1;     /* 0..3 : 32 rows of M each   */
  const int wn    = wave & 1;      /* 0..1 : 64 cols of N each   */
  const int r16   = lane & 15;
  const int khalf = lane >> 4;

  const int m0 = blockIdx.y * BM;
  const int n0 = blockIdx.x * BN;

  /* staging: thread -> row tid>>1 (0..127), 16 contiguous elems at col (tid&1)*16 */
  const int s_row = tid >> 1;
  const int s_col = (tid & 1) * 16;
  const bf16* Ag  = A + (long)(m0 + s_row) * lda + s_col;
  int wn_g = n0 + s_row;
  if (wn_g >= N) wn_g = N - 1;   /* clamp: junk row feeds only unstored columns */
  const bf16* Wg  = W + (long)wn_g * ldw + s_col;
  const int s_lds = s_row * STRD + s_col;

  /* LDS byte offsets of this thread's staging slots (aperture low 32 bits) */
  const unsigned ldsA = (unsigned)(size_t)&As[0][s_lds];
  const unsigned ldsW = (unsigned)(size_t)&Ws[0][s_lds];

  f32x8 acc[2][4];
#pragma unroll
  for (int i = 0; i < 2; ++i)
#pragma unroll
    for (int j = 0; j < 4; ++j)
      acc[i][j] = 0.0f;

  /* issue the 4 async B128 copies staging K-tile k0 into buffer p */
  auto stage = [&](int p, int k0) {
    const unsigned po = (unsigned)(p * (int)sizeof(bf16) * LTILE);
    async_b128(ldsA + po,      Ag + k0);
    async_b128(ldsA + po + 16, Ag + k0 + 8);
    async_b128(ldsW + po,      Wg + k0);
    async_b128(ldsW + po + 16, Wg + k0 + 8);
  };

  /* fragment compute from LDS buffer p (ISA 7.12.2 wave32 bf16 layouts) */
  auto compute = [&](int p) {
    bf16x16 af[2], wf[4];
#pragma unroll
    for (int mf = 0; mf < 2; ++mf) {
      int r = (wm * 32 + mf * 16 + r16) * STRD;
      bf16x8 lo = *(const bf16x8*)&As[p][r + khalf * 8];
      bf16x8 hi = *(const bf16x8*)&As[p][r + 16 + khalf * 8];
      af[mf] = __builtin_shufflevector(lo, hi, 0,1,2,3,4,5,6,7,8,9,10,11,12,13,14,15);
    }
#pragma unroll
    for (int nf = 0; nf < 4; ++nf) {
      int r = (wn * 64 + nf * 16 + r16) * STRD;
      bf16x8 lo = *(const bf16x8*)&Ws[p][r + khalf * 8];
      bf16x8 hi = *(const bf16x8*)&Ws[p][r + 16 + khalf * 8];
      wf[nf] = __builtin_shufflevector(lo, hi, 0,1,2,3,4,5,6,7,8,9,10,11,12,13,14,15);
    }
#pragma unroll
    for (int mf = 0; mf < 2; ++mf)
#pragma unroll
      for (int nf = 0; nf < 4; ++nf)
        acc[mf][nf] = __builtin_amdgcn_wmma_f32_16x16x32_bf16(
            false, af[mf], false, wf[nf], (short)0, acc[mf][nf], false, false);
  };

  /* prologue: DMA tiles 0 and 1; wait until tile 0 landed (<=4 outstanding) */
  stage(0, 0);
  stage(1, BK);
  wait_async4();
  __syncthreads();

  int p = 0;
  for (int kT = 2 * BK; kT < K; kT += BK) {
    int p2 = p + 2; if (p2 >= NSTAGE) p2 -= NSTAGE;
    stage(p2, kT);                        /* tile t+2: two compute phases of cover */
    __builtin_prefetch(Ag + kT, 0, 1);    /* warm GL2 ahead of the DMA engine */
    compute(p);
    wait_async4();                        /* tile t+1 landed ... */
    __syncthreads();                      /* ... collectively */
    if (++p == NSTAGE) p = 0;
  }
  /* drain: two tiles remain ready/landing */
  compute(p);
  if (++p == NSTAGE) p = 0;
  wait_async0();
  __syncthreads();
  compute(p);

  /* epilogue: lane holds M = v + khalf*8, N = r16 (per frag) */
#pragma unroll
  for (int mf = 0; mf < 2; ++mf) {
#pragma unroll
    for (int nf = 0; nf < 4; ++nf) {
      int gm = m0 + wm * 32 + mf * 16 + khalf * 8;
      int gn = n0 + wn * 64 + nf * 16 + r16;
      if (gn >= N) continue;
      float bb = 0.0f;
      if (bias0) bb += bias0[gn];
      if (bias1) bb += bias1[gn];
#pragma unroll
      for (int v = 0; v < 8; ++v) {
        float val = acc[mf][nf][v] + bb;
        if (flags & FLAG_RELU) val = fmaxf(val, 0.0f);
        C[(long)(gm + v) * ldc + gn] = val;
      }
    }
  }
}

/* ---------------- elementwise kernels ---------------- */

__global__ void conv_f32_bf16(const float* __restrict__ s, bf16* __restrict__ d, long n) {
  long i = (long)blockIdx.x * blockDim.x + threadIdx.x;
  long st = (long)gridDim.x * blockDim.x;
  for (; i < n; i += st) d[i] = (bf16)s[i];
}

/* dst[r, 0:Ka] = Wa[r]; dst[r, Ka:Ka+Kb] = Wb[r]   (f32 -> bf16) */
__global__ void pack_cat_bf16(const float* __restrict__ Wa, int Ka,
                              const float* __restrict__ Wb, int Kb,
                              bf16* __restrict__ dst, long rows) {
  long tot = rows * (long)(Ka + Kb);
  long i = (long)blockIdx.x * blockDim.x + threadIdx.x;
  long st = (long)gridDim.x * blockDim.x;
  for (; i < tot; i += st) {
    long row = i / (Ka + Kb);
    int col  = (int)(i - row * (Ka + Kb));
    float v = (col < Ka) ? Wa[row * Ka + col] : Wb[row * Kb + (col - Ka)];
    dst[i] = (bf16)v;
  }
}

__global__ void zero_u32(unsigned int* __restrict__ p, long n) {
  long i = (long)blockIdx.x * blockDim.x + threadIdx.x;
  long st = (long)gridDim.x * blockDim.x;
  for (; i < n; i += st) p[i] = 0u;
}

/* xh[:, 0:1024] = inputs ; xh[:, 1024:1280] = slot embedding (bf16) */
__global__ void build_x_kernel(const float* __restrict__ inputs,
                               const float* __restrict__ start,
                               const float* __restrict__ emb,
                               const int* __restrict__ labels,
                               int slot, bf16* __restrict__ xh) {
  long idx = (long)blockIdx.x * blockDim.x + threadIdx.x;
  if (idx >= (long)BSZ * D0) return;
  int b = (int)(idx / D0);
  int col = (int)(idx % D0);
  float v;
  if (col < DIN) {
    v = inputs[(long)b * DIN + col];
  } else {
    int e = col - DIN;
    if (slot == 0) {
      v = start[e];
    } else {
      int lab = labels[(long)(slot - 1) * BSZ + b];
      v = emb[((long)(slot - 1) * VDIM + lab) * EEMB + e];
    }
  }
  xh[(long)b * XHW + col] = (bf16)v;
}

/* LSTM cell pointwise: gates [B,4H] (i|f|g|o), c in/out, h -> bf16 (strided) + f32(opt) */
__global__ void lstm_pointwise(const float* __restrict__ gates,
                               float* __restrict__ c,
                               float* __restrict__ hf,   /* may be null */
                               bf16* __restrict__ hb, int ldh) {
  long idx = (long)blockIdx.x * blockDim.x + threadIdx.x;
  if (idx >= (long)BSZ * HDIM) return;
  int b = (int)(idx >> 9);           /* HDIM == 512 */
  int h = (int)(idx & (HDIM - 1));
  const float* g = gates + (long)b * G4H;
  float ig = sigmoidf_(g[h]);
  float fg = sigmoidf_(g[HDIM + h]);
  float gg = tanhf(g[2 * HDIM + h]);
  float og = sigmoidf_(g[3 * HDIM + h]);
  float cn = fg * c[idx] + ig * gg;
  c[idx] = cn;
  float hn = og * tanhf(cn);
  if (hf) hf[idx] = hn;
  hb[(long)b * ldh + h] = (bf16)hn;
}

/* highway: cur = sig(ng)*h0 + (1-sig(ng))*xl  -> bf16 (strided dest) */
__global__ void highway_pointwise(const float* __restrict__ ng,
                                  const float* __restrict__ xl,
                                  const float* __restrict__ h0f,
                                  bf16* __restrict__ dst, int ldd) {
  long idx = (long)blockIdx.x * blockDim.x + threadIdx.x;
  if (idx >= (long)BSZ * HDIM) return;
  int b = (int)(idx >> 9);
  int h = (int)(idx & (HDIM - 1));
  float g = sigmoidf_(ng[idx]);
  dst[(long)b * ldd + h] = (bf16)(g * h0f[idx] + (1.0f - g) * xl[idx]);
}

/* ---------------- host orchestration ---------------- */

static inline void launch_gemm(const bf16* A, int lda, const bf16* W, int ldw,
                               float* C, int ldc, int N, int K,
                               const float* b0, const float* b1, int flags,
                               hipStream_t s) {
  dim3 grid((N + BN - 1) / BN, BSZ / BM);
  gemm_bf16_nt<<<grid, 256, 0, s>>>(A, lda, W, ldw, C, ldc, N, K, b0, b1, flags);
}

extern "C" void kernel_launch(void* const* d_in, const int* in_sizes, int n_in,
                              void* d_out, int out_size, void* d_ws, size_t ws_size,
                              hipStream_t stream) {
  (void)in_sizes; (void)n_in; (void)out_size;

  const float* inputs = (const float*)d_in[0];
  const int*   labels = (const int*)  d_in[1];
  const float* start  = (const float*)d_in[2];
  const float* emb    = (const float*)d_in[3];
  const float* W_ih0  = (const float*)d_in[4];
  const float* W_hh0  = (const float*)d_in[5];
  const float* b_ih0  = (const float*)d_in[6];
  const float* b_hh0  = (const float*)d_in[7];
  const float* hwN_W0 = (const float*)d_in[8];
  const float* hwN_b0 = (const float*)d_in[9];
  const float* hwL_W0 = (const float*)d_in[10];
  const float* W_ih1  = (const float*)d_in[11];
  const float* W_hh1  = (const float*)d_in[12];
  const float* b_ih1  = (const float*)d_in[13];
  const float* b_hh1  = (const float*)d_in[14];
  /* d_in[15..17] = hwN_W1/hwN_b1/hwL_W1: dead code in reference, skipped */
  const float* sh_W   = (const float*)d_in[18];
  const float* sh_b   = (const float*)d_in[19];
  const float* pred_W = (const float*)d_in[20];
  const float* pred_b = (const float*)d_in[21];
  float* out = (float*)d_out;

  char* ws = (char*)d_ws;
  size_t off = 0;
  auto take = [&](long elems, size_t esz) -> void* {
    void* p = ws + off;
    off += ((size_t)elems * esz + 255) & ~(size_t)255;
    return p;
  };

  /* packed bf16 weights (rebuilt every call; deterministic) */
  bf16* wcat0 = (bf16*)take((long)NSLOT * G4H * XHW, 2);   /* [W_ih0 | W_hh0] K=1792 */
  bf16* wcat1 = (bf16*)take((long)NSLOT * G4H * CHW, 2);   /* [W_ih1 | W_hh1] K=1024 */
  bf16* hwn0b = (bf16*)take((long)NSLOT * HDIM * XHW, 2);
  bf16* hwl0b = (bf16*)take((long)NSLOT * HDIM * D0, 2);
  bf16* shwb  = (bf16*)take((long)NSLOT * SHDIM * HDIM, 2);
  bf16* predb = (bf16*)take((long)NSLOT * VDIM * SHDIM, 2);
  /* activations */
  bf16* xh    = (bf16*)take((long)BSZ * XHW, 2);   /* [x | h0] */
  bf16* curh  = (bf16*)take((long)BSZ * CHW, 2);   /* [cur | h1] */
  bf16* hidb  = (bf16*)take((long)BSZ * SHDIM, 2);
  float* gatesF = (float*)take((long)BSZ * G4H, 4); /* reused as scratch */
  float* c0f  = (float*)take((long)BSZ * HDIM, 4);
  float* c1f  = (float*)take((long)BSZ * HDIM, 4);
  float* h0f  = (float*)take((long)BSZ * HDIM, 4);
  if (off > ws_size) return;  /* workspace too small */

  float* gbuf  = gatesF;                       /* highway gate pre-acts */
  float* xlbuf = gatesF + (long)BSZ * HDIM;    /* hwL linear path       */
  float* hidf  = gatesF;                       /* slot-hidden post-ReLU   */

  const int CT = 256;
  auto gsz = [&](long n) {
    long g = (n + CT - 1) / CT; if (g > 1048576L) g = 1048576L; return (unsigned)g;
  };
  auto conv = [&](const float* s, bf16* d, long n) {
    conv_f32_bf16<<<gsz(n), CT, 0, stream>>>(s, d, n);
  };

  /* weight packing / conversion */
  pack_cat_bf16<<<gsz((long)NSLOT * G4H * XHW), CT, 0, stream>>>(
      W_ih0, D0, W_hh0, HDIM, wcat0, (long)NSLOT * G4H);
  pack_cat_bf16<<<gsz((long)NSLOT * G4H * CHW), CT, 0, stream>>>(
      W_ih1, HDIM, W_hh1, HDIM, wcat1, (long)NSLOT * G4H);
  conv(hwN_W0, hwn0b, (long)NSLOT * HDIM * XHW);
  conv(hwL_W0, hwl0b, (long)NSLOT * HDIM * D0);
  conv(sh_W,   shwb,  (long)NSLOT * SHDIM * HDIM);
  conv(pred_W, predb, (long)NSLOT * VDIM * SHDIM);

  auto zero = [&](void* p, long bytes) {
    long n = bytes / 4;
    zero_u32<<<gsz(n), CT, 0, stream>>>((unsigned int*)p, n);
  };
  zero(xh,   (long)BSZ * XHW * 2);   /* h0 region must start at 0 */
  zero(curh, (long)BSZ * CHW * 2);   /* h1 region must start at 0 */
  zero(c0f,  (long)BSZ * HDIM * 4);
  zero(c1f,  (long)BSZ * HDIM * 4);

  const unsigned gBH = gsz((long)BSZ * HDIM);
  const unsigned gBX = gsz((long)BSZ * D0);

  for (int i = 0; i < NSLOT; ++i) {
    /* x = concat(inputs, cur_emb) -> xh[:, :D0]; xh[:, D0:] holds prev h0 */
    build_x_kernel<<<gBX, CT, 0, stream>>>(inputs, start, emb, labels, i, xh);

    /* layer-0 gates = [x|h0] @ [W_ih0|W_hh0]^T + b_ih0 + b_hh0   (one GEMM) */
    launch_gemm(xh, XHW, wcat0 + (long)i * G4H * XHW, XHW, gatesF, G4H,
                G4H, XHW, b_ih0 + (long)i * G4H, b_hh0 + (long)i * G4H, 0, stream);
    /* h0,c0 update; h0(bf16) -> xh[:, D0:], h0(f32) -> h0f */
    lstm_pointwise<<<gBH, CT, 0, stream>>>(gatesF, c0f, h0f, xh + D0, XHW);

    /* highway: gate = sig([x,h0]@hwN^T + b) ; xl = x@hwL^T */
    launch_gemm(xh, XHW, hwn0b + (long)i * HDIM * XHW, XHW, gbuf, HDIM,
                HDIM, XHW, hwN_b0 + (long)i * HDIM, nullptr, 0, stream);
    launch_gemm(xh, XHW, hwl0b + (long)i * HDIM * D0, D0, xlbuf, HDIM,
                HDIM, D0, nullptr, nullptr, 0, stream);
    highway_pointwise<<<gBH, CT, 0, stream>>>(gbuf, xlbuf, h0f, curh, CHW);

    /* layer-1 gates = [cur|h1] @ [W_ih1|W_hh1]^T + b_ih1 + b_hh1 (one GEMM) */
    launch_gemm(curh, CHW, wcat1 + (long)i * G4H * CHW, CHW, gatesF, G4H,
                G4H, CHW, b_ih1 + (long)i * G4H, b_hh1 + (long)i * G4H, 0, stream);
    /* h1,c1 update; h1(bf16) -> curh[:, H:] */
    lstm_pointwise<<<gBH, CT, 0, stream>>>(gatesF, c1f, nullptr, curh + HDIM, CHW);
    /* layer-1 highway output unused by the reference: skipped */

    /* hidden = relu(h1@sh_W^T + sh_b) */
    launch_gemm(curh + HDIM, CHW, shwb + (long)i * SHDIM * HDIM, HDIM, hidf, SHDIM,
                SHDIM, HDIM, sh_b + (long)i * SHDIM, nullptr, FLAG_RELU, stream);
    conv(hidf, hidb, (long)BSZ * SHDIM);

    /* logits[i] = hidden@pred_W^T + pred_b */
    launch_gemm(hidb, SHDIM, predb + (long)i * VDIM * SHDIM, SHDIM,
                out + (long)i * BSZ * VDIM, VDIM,
                VDIM, SHDIM, pred_b + (long)i * VDIM, nullptr, 0, stream);
  }
}
